// EdgeGAT_86294482911651
// MI455X (gfx1250) — matrix operand order repaired and checked by reading, not compiled, not run on hardware
//
#include <hip/hip_runtime.h>
#include <hip/hip_bf16.h>

typedef __attribute__((ext_vector_type(16))) __bf16 v16bf;
typedef __attribute__((ext_vector_type(8)))  float  v8f;

// ---------------- ordered-float encoding for atomic max ----------------
__device__ __forceinline__ unsigned fenc(float f) {
    unsigned u = __float_as_uint(f);
    return (u & 0x80000000u) ? ~u : (u | 0x80000000u);
}
__device__ __forceinline__ float fdec(unsigned u) {
    unsigned b = (u & 0x80000000u) ? (u & 0x7fffffffu) : ~u;
    return __uint_as_float(b);
}
__device__ __forceinline__ float lrelu(float x) { return x > 0.f ? x : 0.2f * x; }

// ---------------- WMMA fragment helpers (CDNA5 wave32 layouts) ----------------
// Pack 16 f32 (four float4 runs) into the 16x32 bf16 A fragment for this lane.
__device__ __forceinline__ v16bf pack_a(float4 q0, float4 q1, float4 q2, float4 q3) {
    v16bf a;
    a[0]  = (__bf16)q0.x; a[1]  = (__bf16)q0.y; a[2]  = (__bf16)q0.z; a[3]  = (__bf16)q0.w;
    a[4]  = (__bf16)q1.x; a[5]  = (__bf16)q1.y; a[6]  = (__bf16)q1.z; a[7]  = (__bf16)q1.w;
    a[8]  = (__bf16)q2.x; a[9]  = (__bf16)q2.y; a[10] = (__bf16)q2.z; a[11] = (__bf16)q2.w;
    a[12] = (__bf16)q3.x; a[13] = (__bf16)q3.y; a[14] = (__bf16)q3.z; a[15] = (__bf16)q3.w;
    return a;
}

// A (16x32 bf16): lanes 0-15 rows M=0..15 hold K=0..7,16..23 ; lanes 16-31 hold K=8..15,24..31.
// Per lane: two contiguous 8-float runs -> 4x global_load_b128.
__device__ __forceinline__ v16bf load_a_frag(const float* __restrict__ A, int lda, int rowBase, int k0) {
    int lane = threadIdx.x & 31;
    int r = rowBase + (lane & 15);
    int kh = (lane >> 4) & 1;
    const float* p0 = A + (long long)r * lda + k0 + kh * 8;
    const float4* q = (const float4*)p0;
    const float4* q16 = (const float4*)(p0 + 16);
    return pack_a(q[0], q[1], q16[0], q16[1]);
}

// B (32x16 bf16, row-major source [K, ldb]): lanes 0-15 cols N=0..15 hold K=0..7,16..23,
// lanes 16-31 hold K=8..15,24..31. Loads are coalesced across lanes.
__device__ __forceinline__ v16bf load_b_frag(const float* __restrict__ B, int ldb, int k0, int colBase) {
    int lane = threadIdx.x & 31;
    int n = colBase + (lane & 15);
    int kh = (lane >> 4) & 1;
    const float* p = B + (long long)(k0 + 8 * kh) * ldb + n;
    v16bf b;
#pragma unroll
    for (int j = 0; j < 8; ++j) {
        b[j]     = (__bf16)p[(long long)j * ldb];
        b[8 + j] = (__bf16)p[(long long)(16 + j) * ldb];
    }
    return b;
}

// ---------------- generic WMMA GEMM: C[M,Ncols] = A[M,K] @ B[K,Ncols] ----------------
// blockDim.x = 128 (4 waves); each wave -> one 16x16 C tile.
// Requirements (guaranteed by launcher): M % 16 == 0, K % 32 == 0, Ncols % (16*gridDim-cover) == 0.
__global__ void wmma_gemm(const float* __restrict__ A, const float* __restrict__ B,
                          float* __restrict__ C, int K, int Ncols) {
    int wave = threadIdx.x >> 5;
    int rowBase = blockIdx.x * 16;
    int colBase = (blockIdx.y * 4 + wave) * 16;
    v8f acc = {};
    for (int k0 = 0; k0 < K; k0 += 32) {
        v16bf a = load_a_frag(A, K, rowBase, k0);
        v16bf b = load_b_frag(B, Ncols, k0, colBase);
        acc = __builtin_amdgcn_wmma_f32_16x16x32_bf16(false, a, false, b, (short)0, acc, false, false);
    }
    // C/D layout: VGPR r -> row (lane<16 ? r : r+8), col = lane&15
    int lane = threadIdx.x & 31;
    int col = colBase + (lane & 15);
    int rb = (lane >> 4) * 8;
    float* cp = C + (long long)(rowBase + rb) * Ncols + col;
#pragma unroll
    for (int r = 0; r < 8; ++r)
        cp[(long long)r * Ncols] = acc[r];
}

// ---------------- attention logits: a_s[n,h] = <h[n,h,:], att_s[h,:]> ----------------
__global__ void att_logits(const float* __restrict__ h, const float* __restrict__ att_s,
                           const float* __restrict__ att_d, float* __restrict__ as,
                           float* __restrict__ ad, int n, int heads, int C) {
    int i = blockIdx.x * blockDim.x + threadIdx.x;
    if (i >= n * heads) return;
    int node = i / heads, hd = i % heads;
    const float* hp = h + (long long)node * heads * C + hd * C;
    const float* sp = att_s + hd * C;
    const float* dp = att_d + hd * C;
    float s0 = 0.f, d0 = 0.f;
    for (int c = 0; c < C; c += 4) {
        float4 v = *(const float4*)(hp + c);
        float4 sv = *(const float4*)(sp + c);
        float4 dv = *(const float4*)(dp + c);
        s0 += v.x * sv.x + v.y * sv.y + v.z * sv.z + v.w * sv.w;
        d0 += v.x * dv.x + v.y * dv.y + v.z * dv.z + v.w * dv.w;
    }
    as[i] = s0; ad[i] = d0;
}

// ---------------- fills ----------------
__global__ void fill_f32(float* p, float v, long long n) {
    long long i = (long long)blockIdx.x * blockDim.x + threadIdx.x;
    if (i < n) p[i] = v;
}
__global__ void fill_u32(unsigned* p, unsigned v, long long n) {
    long long i = (long long)blockIdx.x * blockDim.x + threadIdx.x;
    if (i < n) p[i] = v;
}

// ---------------- edge softmax passes (self loops generated virtually) ----------------
__global__ void edge_max(const int* __restrict__ ei, int E, int n, const float* __restrict__ as,
                         const float* __restrict__ ad, unsigned* __restrict__ m_u, int heads) {
    long long i = (long long)blockIdx.x * blockDim.x + threadIdx.x;
    if (i >= (long long)E + n) return;
    int src = (i < E) ? ei[i] : (int)(i - E);
    int dst = (i < E) ? ei[(long long)E + i] : (int)(i - E);
    for (int h = 0; h < heads; ++h) {
        float e = lrelu(as[src * heads + h] + ad[dst * heads + h]);
        atomicMax(&m_u[dst * heads + h], fenc(e));
    }
}

__global__ void edge_sum(const int* __restrict__ ei, int E, int n, const float* __restrict__ as,
                         const float* __restrict__ ad, const unsigned* __restrict__ m_u,
                         float* __restrict__ ssum, int heads) {
    long long i = (long long)blockIdx.x * blockDim.x + threadIdx.x;
    if (i >= (long long)E + n) return;
    int src = (i < E) ? ei[i] : (int)(i - E);
    int dst = (i < E) ? ei[(long long)E + i] : (int)(i - E);
    for (int h = 0; h < heads; ++h) {
        float e = lrelu(as[src * heads + h] + ad[dst * heads + h]);
        atomicAdd(&ssum[dst * heads + h], __expf(e - fdec(m_u[dst * heads + h])));
    }
}

// one wave per edge; lanes stride over C features of each head
__global__ void edge_scatter(const int* __restrict__ ei, int E, int n, const float* __restrict__ as,
                             const float* __restrict__ ad, const unsigned* __restrict__ m_u,
                             const float* __restrict__ ssum, const float* __restrict__ hsrc,
                             float* __restrict__ out, int heads, int C) {
    long long tid = (long long)blockIdx.x * blockDim.x + threadIdx.x;
    int lane = (int)(tid & 31);
    long long e = tid >> 5;
    if (e >= (long long)E + n) return;
    int src = (e < E) ? ei[e] : (int)(e - E);
    int dst = (e < E) ? ei[(long long)E + e] : (int)(e - E);
    int F = heads * C;
    for (int h = 0; h < heads; ++h) {
        float ev = lrelu(as[src * heads + h] + ad[dst * heads + h]);
        float w = __expf(ev - fdec(m_u[dst * heads + h])) / (ssum[dst * heads + h] + 1e-16f);
        for (int c = lane; c < C; c += 32)
            atomicAdd(&out[(long long)dst * F + h * C + c], w * hsrc[(long long)src * F + h * C + c]);
    }
}

__global__ void bias_act(float* __restrict__ h, const float* __restrict__ b, long long n, int F, int relu) {
    long long i = (long long)blockIdx.x * blockDim.x + threadIdx.x;
    if (i >= n * F) return;
    float v = h[i] + b[i % F];
    if (relu) v = fmaxf(v, 0.f);
    h[i] = v;
}

// ---------------- edge MLP: [E,160] @ Wm1[160,64] -> relu -> @ Wm2[64,2] ----------------
// block = 128 threads (4 waves), 16 edges per block; wave w -> hidden cols [16w, 16w+16).
// Requires E % 16 == 0 (E = 1e6).
__device__ __forceinline__ const float* mlp_src(const float* __restrict__ h2, int s, int d,
                                                const float* __restrict__ eattr, int er, int k) {
    if (k < 64)  return h2 + (long long)s * 64 + k;
    if (k < 128) return h2 + (long long)d * 64 + (k - 64);
    return eattr + (long long)er * 32 + (k - 128);
}

__global__ void edge_mlp(const float* __restrict__ h2, const float* __restrict__ eattr,
                         const int* __restrict__ esrc, const int* __restrict__ edst,
                         const float* __restrict__ Wm1, const float* __restrict__ bm1,
                         const float* __restrict__ Wm2, const float* __restrict__ bm2,
                         float* __restrict__ out, int E) {
    __shared__ float hid[16][64];
    int wave = threadIdx.x >> 5;
    int lane = threadIdx.x & 31;
    int ebase = blockIdx.x * 16;
    int er = ebase + (lane & 15);
    int s = esrc[er], d = edst[er];
    int kh = (lane >> 4) & 1;
    int colBase = wave * 16;
    v8f acc = {};
#pragma unroll
    for (int k0 = 0; k0 < 160; k0 += 32) {
        // Two 8-float runs per lane; runs are 8-aligned and never cross the 64/128 segment
        // boundaries (k0 steps by 32), so each run is one B128-pair from a single source.
        int kA = k0 + kh * 8;
        int kB = k0 + 16 + kh * 8;
        const float4* pa = (const float4*)mlp_src(h2, s, d, eattr, er, kA);
        const float4* pb = (const float4*)mlp_src(h2, s, d, eattr, er, kB);
        v16bf a = pack_a(pa[0], pa[1], pb[0], pb[1]);
        v16bf b = load_b_frag(Wm1, 64, k0, colBase);
        acc = __builtin_amdgcn_wmma_f32_16x16x32_bf16(false, a, false, b, (short)0, acc, false, false);
    }
    int col = colBase + (lane & 15);
    int rb = (lane >> 4) * 8;
#pragma unroll
    for (int r = 0; r < 8; ++r)
        hid[rb + r][col] = fmaxf(acc[r] + bm1[col], 0.f);
    __syncthreads();
    if (threadIdx.x < 32) {
        int el = threadIdx.x >> 1;
        int j = threadIdx.x & 1;
        float sum = bm2[j];
#pragma unroll 8
        for (int c = 0; c < 64; ++c) sum += hid[el][c] * Wm2[c * 2 + j];
        out[(long long)(ebase + el) * 2 + j] = sum;
    }
}

// ---------------- launcher ----------------
extern "C" void kernel_launch(void* const* d_in, const int* in_sizes, int n_in,
                              void* d_out, int out_size, void* d_ws, size_t ws_size,
                              hipStream_t stream) {
    const float* x         = (const float*)d_in[0];
    const int*   ei        = (const int*)d_in[1];
    const float* eattr     = (const float*)d_in[2];
    const int*   esrc      = (const int*)d_in[3];
    const int*   edst      = (const int*)d_in[4];
    const float* W1        = (const float*)d_in[5];
    const float* att_src1  = (const float*)d_in[6];
    const float* att_dst1  = (const float*)d_in[7];
    const float* b1        = (const float*)d_in[8];
    const float* W2        = (const float*)d_in[9];
    const float* att_src2  = (const float*)d_in[10];
    const float* att_dst2  = (const float*)d_in[11];
    const float* b2        = (const float*)d_in[12];
    const float* Wm1       = (const float*)d_in[13];
    const float* bm1       = (const float*)d_in[14];
    const float* Wm2       = (const float*)d_in[15];
    const float* bm2       = (const float*)d_in[16];

    const int N = in_sizes[0] / 64;   // F_IN = 64; N = 100000 (multiple of 16)
    const int E = in_sizes[3];        // E = 1e6 (multiple of 16)
    const long long Ep = (long long)E + N;

    // ---- workspace carve-up ----
    char* ws = (char*)d_ws;
    float*    h1_lin = (float*)ws;                                  // N*128
    float*    h1     = (float*)(ws + (size_t)N * 128 * 4);          // N*128
    char*     sm     = ws + (size_t)N * 128 * 4 * 2;
    float*    a_s1   = (float*)sm;              sm += (size_t)N * 2 * 4;
    float*    a_d1   = (float*)sm;              sm += (size_t)N * 2 * 4;
    unsigned* m1_u   = (unsigned*)sm;           sm += (size_t)N * 2 * 4;
    float*    s1     = (float*)sm;              sm += (size_t)N * 2 * 4;
    float*    a_s2   = (float*)sm;              sm += (size_t)N * 4;
    float*    a_d2   = (float*)sm;              sm += (size_t)N * 4;
    unsigned* m2_u   = (unsigned*)sm;           sm += (size_t)N * 4;
    float*    s2     = (float*)sm;              sm += (size_t)N * 4;
    // h1_lin is dead after layer-1 aggregation: reuse its region for layer 2
    float*    h2_lin = h1_lin;                                      // N*64
    float*    h2     = h1_lin + (size_t)N * 64;                     // N*64

    const int TB = 256;
    dim3 blk128(128);

    // ===== Layer 1 =====
    {
        dim3 grid(N / 16, 128 / 64);
        wmma_gemm<<<grid, blk128, 0, stream>>>(x, W1, h1_lin, 64, 128);
    }
    att_logits<<<(N * 2 + TB - 1) / TB, TB, 0, stream>>>(h1_lin, att_src1, att_dst1, a_s1, a_d1, N, 2, 64);
    fill_u32<<<(unsigned)((N * 2LL + TB - 1) / TB), TB, 0, stream>>>(m1_u, 0u, (long long)N * 2);
    fill_f32<<<(unsigned)((N * 2LL + TB - 1) / TB), TB, 0, stream>>>(s1, 0.f, (long long)N * 2);
    fill_f32<<<(unsigned)(((long long)N * 128 + TB - 1) / TB), TB, 0, stream>>>(h1, 0.f, (long long)N * 128);
    edge_max<<<(unsigned)((Ep + TB - 1) / TB), TB, 0, stream>>>(ei, E, N, a_s1, a_d1, m1_u, 2);
    edge_sum<<<(unsigned)((Ep + TB - 1) / TB), TB, 0, stream>>>(ei, E, N, a_s1, a_d1, m1_u, s1, 2);
    edge_scatter<<<(unsigned)((Ep * 32 + TB - 1) / TB), TB, 0, stream>>>(ei, E, N, a_s1, a_d1, m1_u, s1,
                                                                          h1_lin, h1, 2, 64);
    bias_act<<<(unsigned)(((long long)N * 128 + TB - 1) / TB), TB, 0, stream>>>(h1, b1, N, 128, 1);

    // ===== Layer 2 =====
    {
        dim3 grid(N / 16, 1);
        wmma_gemm<<<grid, blk128, 0, stream>>>(h1, W2, h2_lin, 128, 64);
    }
    att_logits<<<(N + TB - 1) / TB, TB, 0, stream>>>(h2_lin, att_src2, att_dst2, a_s2, a_d2, N, 1, 64);
    fill_u32<<<(unsigned)((N + TB - 1LL) / TB), TB, 0, stream>>>(m2_u, 0u, (long long)N);
    fill_f32<<<(unsigned)((N + TB - 1LL) / TB), TB, 0, stream>>>(s2, 0.f, (long long)N);
    fill_f32<<<(unsigned)(((long long)N * 64 + TB - 1) / TB), TB, 0, stream>>>(h2, 0.f, (long long)N * 64);
    edge_max<<<(unsigned)((Ep + TB - 1) / TB), TB, 0, stream>>>(ei, E, N, a_s2, a_d2, m2_u, 1);
    edge_sum<<<(unsigned)((Ep + TB - 1) / TB), TB, 0, stream>>>(ei, E, N, a_s2, a_d2, m2_u, s2, 1);
    edge_scatter<<<(unsigned)((Ep * 32 + TB - 1) / TB), TB, 0, stream>>>(ei, E, N, a_s2, a_d2, m2_u, s2,
                                                                          h2_lin, h2, 1, 64);
    bias_act<<<(unsigned)(((long long)N * 64 + TB - 1) / TB), TB, 0, stream>>>(h2, b2, N, 64, 0);

    // ===== Edge MLP =====
    edge_mlp<<<E / 16, blk128, 0, stream>>>(h2, eattr, esrc, edst, Wm1, bm1, Wm2, bm2,
                                            (float*)d_out, E);
}
